// VarietyAdapterHead_48730698940499
// MI455X (gfx1250) — matrix operand order repaired and checked by reading, not compiled, not run on hardware
//
#include <hip/hip_runtime.h>
#include <hip/hip_bf16.h>

typedef __attribute__((ext_vector_type(16))) __bf16 v16bf;
typedef __attribute__((ext_vector_type(8)))  __bf16 v8bf;
typedef __attribute__((ext_vector_type(8)))  float  v8f;

#define B_ 4096
#define T_ 16
#define H_ 768
#define A_ 128
#define V_ 16
#define L_ 2

// workspace layout (bytes)
#define OFF_COUNTS 0            // 16 ints
#define OFF_LISTS  256          // 16*4096 ints = 262144 B
#define OFF_WDT    262656       // 16*768*128 bf16 = 3145728 B  (layout [v][a][k])
#define OFF_WUT    3408384      // 16*768*128 bf16 = 3145728 B  (layout [v][h][a])
// total ~6.26 MB

// Branch-free tanh: prefer the gfx1250 hardware V_TANH_F32 transcendental.
__device__ __forceinline__ float fast_tanh(float z) {
#if defined(__has_builtin)
#if __has_builtin(__builtin_amdgcn_tanh_f32)
    return __builtin_amdgcn_tanh_f32(z);
#else
    // branchless, overflow-safe: tanh(|z|) = (1-e)/(1+e), e = exp(-2|z|) in (0,1]
    float az = __builtin_fabsf(z);
    float e  = __expf(-2.0f * az);
    float t  = (1.0f - e) / (1.0f + e);
    return (z < 0.0f) ? -t : t;
#endif
#else
    float az = __builtin_fabsf(z);
    float e  = __expf(-2.0f * az);
    float t  = (1.0f - e) / (1.0f + e);
    return (z < 0.0f) ? -t : t;
#endif
}

// ---------------------------------------------------------------------------
// Kernel 0: convert f32 weights to bf16 with transpose so WMMA B-fragments are
// contiguous 32B vector loads.
//   WdT[v][a][k] = Wd[v][k][a]      (down-proj B: K=h-dim, N=a-dim)
//   WuT[v][h][a] = Wu[v][a][h]      (up-proj   B: K=a-dim, N=h-dim)
// ---------------------------------------------------------------------------
__global__ void convert_weights(const float* __restrict__ Wd,
                                const float* __restrict__ Wu,
                                __bf16* __restrict__ wdT,
                                __bf16* __restrict__ wuT) {
    const int N1 = V_ * H_ * A_;
    int idx = blockIdx.x * blockDim.x + threadIdx.x;
    if (idx < N1) {
        int v = idx / (H_ * A_);
        int r = idx - v * (H_ * A_);
        int a = r / H_;
        int k = r - a * H_;
        wdT[idx] = (__bf16)Wd[v * (H_ * A_) + k * A_ + a];
    } else {
        int j = idx - N1;
        if (j < N1) {
            int v = j / (H_ * A_);
            int r = j - v * (H_ * A_);
            int h = r / A_;
            int a = r - h * A_;
            wuT[j] = (__bf16)Wu[v * (A_ * H_) + a * H_ + h];
        }
    }
}

// ---------------------------------------------------------------------------
// Kernels 1/2: bucket rows by variety. Atomic order is nondeterministic but
// each row's result is independent of its tile slot, so output is
// bit-deterministic.
// ---------------------------------------------------------------------------
__global__ void zero_counts(int* counts) {
    if (threadIdx.x < V_) counts[threadIdx.x] = 0;
}

__global__ void scatter_rows(const int* __restrict__ vid,
                             int* __restrict__ counts,
                             int* __restrict__ lists) {
    int b = blockIdx.x * blockDim.x + threadIdx.x;
    if (b < B_) {
        int v = vid[b] & (V_ - 1);
        int pos = atomicAdd(&counts[v], 1);
        lists[v * B_ + pos] = b;
    }
}

// ---------------------------------------------------------------------------
// Main kernel: one block = one 16-row tile of one variety. 256 threads = 8
// waves. Wave w owns N-columns [16w,16w+16) of the down-proj and 6 N-tiles of
// the up-proj. All matrix math via v_wmma_f32_16x16x32_bf16.
// ---------------------------------------------------------------------------
__launch_bounds__(256)
__global__ void adapter_main(const float* __restrict__ X,     // last_hidden (B,T,H)
                             const float* __restrict__ bd,    // (V,A)
                             const float* __restrict__ bu,    // (V,H)
                             const float* __restrict__ Wc,    // (L,H)
                             const float* __restrict__ bc,    // (L,)
                             const __bf16* __restrict__ wdT,  // [v][a][k]
                             const __bf16* __restrict__ wuT,  // [v][h][a]
                             const int* __restrict__ counts,
                             const int* __restrict__ lists,
                             float* __restrict__ out) {
    __shared__ __bf16 xs[16 * H_];   // 24 KB  x tile, bf16
    __shared__ __bf16 hs[16 * A_];   //  4 KB  gelu(down) tile, bf16
    __shared__ float  us[16 * H_];   // 48 KB  adapted tile, f32
    __shared__ int    rows_s[16];

    const int v    = blockIdx.x >> 8;
    const int tile = blockIdx.x & 255;
    const int cnt  = counts[v];
    if (tile * 16 >= cnt) return;    // uniform across block: barrier-safe

    const int tid  = threadIdx.x;
    const int wave = tid >> 5;
    const int lane = tid & 31;
    const int hi   = lane >> 4;      // K-half selector in A/B fragments
    const int lm   = lane & 15;      // row (A) / column (B,C,D) within tile

    if (tid < 16) {
        int pos = tile * 16 + tid;
        rows_s[tid] = (pos < cnt) ? lists[v * B_ + pos] : -1;
    }
    __syncthreads();

    // ---- gather x rows -> bf16 LDS (padding rows -> 0) ----
    for (int idx = tid; idx < 16 * H_; idx += 256) {
        int m   = idx / H_;
        int col = idx - m * H_;
        int gr  = rows_s[m];
        float xv = (gr >= 0) ? X[(size_t)gr * (T_ * H_) + col] : 0.0f;
        xs[idx] = (__bf16)xv;
    }
    __syncthreads();

    const __bf16* wdv = wdT + (size_t)v * (H_ * A_);
    const __bf16* wuv = wuT + (size_t)v * (H_ * A_);
    // warm up-proj weights in L2 while down-proj runs (global_prefetch_b8)
    __builtin_prefetch(&wuv[tid * 384], 0, 1);

    // ---- down projection: (16xH) @ (HxA) -> 16x16 per wave ----
    // unroll x4: clause A/B loads ahead of the WMMA pipe instead of a full
    // wait per matrix op.
    v8f acc = {};
    #pragma unroll 4
    for (int kb = 0; kb < H_; kb += 32) {
        // A fragment (16x32 bf16): elt j -> K=kb+hi*16+j (contiguous)
        v8bf a0 = *(const v8bf*)&xs[lm * H_ + kb + hi * 16];
        v8bf a1 = *(const v8bf*)&xs[lm * H_ + kb + hi * 16 + 8];
        v16bf afrag = __builtin_shufflevector(a0, a1,
            0, 1, 2, 3, 4, 5, 6, 7, 8, 9, 10, 11, 12, 13, 14, 15);
        // B fragment (32x16 bf16): elt j -> K=kb+hi*16+j, N=16*wave+lm (contiguous)
        v16bf bfrag = *(const v16bf*)&wdv[(wave * 16 + lm) * H_ + kb + hi * 16];
        acc = __builtin_amdgcn_wmma_f32_16x16x32_bf16(
            false, afrag, false, bfrag, (short)0, acc, false, false);
    }

    // ---- bias + tanh-gelu (hardware v_tanh_f32), write h tile to LDS ----
    {
        int acol = wave * 16 + lm;
        float bias = bd[v * A_ + acol];
        #pragma unroll
        for (int i = 0; i < 8; ++i) {
            int m = i + hi * 8;                        // C/D layout: M = i + 8*hi
            float x = acc[i] + bias;
            float g = 0.5f * x *
                (1.0f + fast_tanh(0.7978845608028654f * (x + 0.044715f * x * x * x)));
            hs[m * A_ + acol] = (__bf16)g;
        }
    }
    __syncthreads();

    // ---- up projection: (16xA) @ (AxH) -> 6 N-tiles per wave ----
    #pragma unroll 1
    for (int t = 0; t < 6; ++t) {
        int nt   = wave * 6 + t;
        int hcol = nt * 16 + lm;
        v8f accu = {};
        #pragma unroll
        for (int kb = 0; kb < A_; kb += 32) {
            v8bf a0 = *(const v8bf*)&hs[lm * A_ + kb + hi * 16];
            v8bf a1 = *(const v8bf*)&hs[lm * A_ + kb + hi * 16 + 8];
            v16bf afrag = __builtin_shufflevector(a0, a1,
                0, 1, 2, 3, 4, 5, 6, 7, 8, 9, 10, 11, 12, 13, 14, 15);
            v16bf bfrag = *(const v16bf*)&wuv[hcol * A_ + kb + hi * 16];
            accu = __builtin_amdgcn_wmma_f32_16x16x32_bf16(
                false, afrag, false, bfrag, (short)0, accu, false, false);
        }
        // residual + bias, adapted tile to LDS f32
        float bias = bu[v * H_ + hcol];
        #pragma unroll
        for (int i = 0; i < 8; ++i) {
            int m  = i + hi * 8;
            int gr = rows_s[m];
            float xv = (gr >= 0) ? X[(size_t)gr * (T_ * H_) + hcol] : 0.0f;
            us[m * H_ + hcol] = xv + accu[i] + bias;
        }
    }
    __syncthreads();

    // ---- final L=2 projection: fixed-order serial dot => deterministic ----
    if (tid < 32) {
        int m  = tid >> 1;
        int lo = tid & 1;
        int gr = rows_s[m];
        if (gr >= 0) {
            float sum = bc[lo];
            const float* wrow = Wc + lo * H_;
            for (int h = 0; h < H_; ++h)
                sum += us[m * H_ + h] * wrow[h];
            out[gr * L_ + lo] = sum;
        }
    }
}

// ---------------------------------------------------------------------------
extern "C" void kernel_launch(void* const* d_in, const int* in_sizes, int n_in,
                              void* d_out, int out_size, void* d_ws, size_t ws_size,
                              hipStream_t stream) {
    const float* X  = (const float*)d_in[0];   // last_hidden (B,T,H)
    // d_in[1] attention_mask: unused by reference
    const int*   vid = (const int*)d_in[2];    // variety_ids (B,)
    const float* Wd = (const float*)d_in[3];   // (V,H,A)
    const float* bd = (const float*)d_in[4];   // (V,A)
    const float* Wu = (const float*)d_in[5];   // (V,A,H)
    const float* bu = (const float*)d_in[6];   // (V,H)
    const float* Wc = (const float*)d_in[7];   // (L,H)
    const float* bc = (const float*)d_in[8];   // (L,)
    float* out = (float*)d_out;

    char* ws = (char*)d_ws;
    int*    counts = (int*)(ws + OFF_COUNTS);
    int*    lists  = (int*)(ws + OFF_LISTS);
    __bf16* wdT    = (__bf16*)(ws + OFF_WDT);
    __bf16* wuT    = (__bf16*)(ws + OFF_WUT);

    const int nconv = 2 * V_ * H_ * A_;
    convert_weights<<<(nconv + 255) / 256, 256, 0, stream>>>(Wd, Wu, wdT, wuT);
    zero_counts<<<1, 32, 0, stream>>>(counts);
    scatter_rows<<<(B_ + 255) / 256, 256, 0, stream>>>(vid, counts, lists);
    adapter_main<<<V_ * 256, 256, 0, stream>>>(X, bd, bu, Wc, bc,
                                               wdT, wuT, counts, lists, out);
}
